// BU_RvNN_8847632630375
// MI455X (gfx1250) — matrix operand order repaired and checked by reading, not compiled, not run on hardware
//
#include <hip/hip_runtime.h>
#include <hip/hip_bf16.h>
#include <stdint.h>

// ---------------------------------------------------------------------------
// BU-RvNN (tree GRU) for MI455X / gfx1250.
// Per layer: gi = X @ w_ih^T + b_ih ; gh = Hprev @ w_hh^T + b_hh ; GRU gates.
// Both GEMMs run on v_wmma_f32_16x16x32_bf16 (K=256 in 8 steps of 32).
// 32-row block tiles: each wave holds 2 row sub-tiles so every pre-swizzled
// B fragment load feeds 2 WMMAs (halves L2 weight traffic vs 16-row tiles).
// ---------------------------------------------------------------------------

typedef __attribute__((ext_vector_type(16))) __bf16 v16bf;
typedef __attribute__((ext_vector_type(8)))  float  v8f;

union V16 { uint4 q[2]; v16bf v; };

__device__ __forceinline__ unsigned short f2bf(float f) {
  unsigned int u = __float_as_uint(f);
  u += 0x7FFFu + ((u >> 16) & 1u);          // round-to-nearest-even
  return (unsigned short)(u >> 16);
}

__device__ __forceinline__ float fsigmoid(float x) {
  return 1.0f / (1.0f + __expf(-x));
}
__device__ __forceinline__ float ftanh(float x) {
  float e = __expf(2.0f * x);
  return (e - 1.0f) / (e + 1.0f);
}

#define HD     256           // hidden size == feature size
#define G3     768           // 3*H gate width
#define JT_N   48            // 768/16 column tiles
#define KT_N   8             // 256/32 K steps
#define FRAG_US 512          // ushorts per B fragment (32 lanes * 16 elems)
#define WB_US  (JT_N * KT_N * FRAG_US)   // 196608 ushorts per weight matrix
#define LMAX_STEPS 40
#define TROWS  32            // rows per block tile

// ---------------- meta: recover (L, M) of layer_nodes on device -------------
// Layer 0 holds exactly T valid entries, rows are front-packed, pad value = N.
// => M = first index k >= T with layer_nodes[k] != N (or LM if none), L = LM/M.
__global__ void meta_init(int* hdr, int LM) {
  if (threadIdx.x == 0 && blockIdx.x == 0) hdr[2] = LM;
}
__global__ void meta_scan(const int* __restrict__ ln, int* hdr,
                          int T, int LM, int Ntot) {
  int stride = gridDim.x * blockDim.x;
  for (int i = blockIdx.x * blockDim.x + threadIdx.x + T; i < LM; i += stride) {
    int v = ln[i];
    if (v != Ntot) {
      // only run-starts hit the atomic (≈ L transitions total)
      if (i == T || ln[i - 1] == Ntot) atomicMin(&hdr[2], i);
    }
  }
}
__global__ void meta_fin(int* hdr, int LM) {
  if (threadIdx.x == 0 && blockIdx.x == 0) {
    int Mv = hdr[2];
    hdr[1] = Mv;
    hdr[0] = LM / Mv;
  }
}

// ---------------- zero the hidden-state buffer ------------------------------
__global__ void zero_f4(float4* p, long long n4) {
  long long i  = (long long)blockIdx.x * blockDim.x + threadIdx.x;
  long long st = (long long)gridDim.x * blockDim.x;
  float4 z = make_float4(0.f, 0.f, 0.f, 0.f);
  for (; i < n4; i += st) p[i] = z;
}

// ---------------- convert + swizzle weights into B-fragment layout ----------
// B fragment (32x16 bf16, K x N): lane l -> col N = l&15, half = l>>4 ;
// element e -> K = kt*32 + half*16 + e   (contiguous k order, same as the A loads).
__global__ void conv_w(const float* __restrict__ wih,
                       const float* __restrict__ whh,
                       unsigned short* __restrict__ wbih,
                       unsigned short* __restrict__ wbhh) {
  int n = blockIdx.x * blockDim.x + threadIdx.x;
  if (n >= 2 * WB_US) return;
  int m = (n >= WB_US) ? 1 : 0;
  int r = n - m * WB_US;
  int e    = r & 15;
  int lane = (r >> 4) & 31;
  int kt   = (r >> 9) & 7;
  int jt   = r >> 12;                      // 0..47
  int j = jt * 16 + (lane & 15);           // output (gate) column, 0..767
  int k = kt * 32 + (lane >> 4) * 16 + e;  // K index, 0..255
  const float* src = m ? whh : wih;        // row-major (768, 256)
  unsigned short* dst = m ? wbhh : wbih;
  dst[r] = f2bf(src[j * 256 + k]);
}

// ---------------- main per-layer kernel -------------------------------------
__global__ __launch_bounds__(256)
void layer_step(int step, const int* hdr,
                const float* __restrict__ inputs,
                const float* __restrict__ b_ih,
                const float* __restrict__ b_hh,
                const int* __restrict__ layer_nodes,
                const int* __restrict__ child_idx,
                const int* __restrict__ child_cnt,
                const unsigned short* __restrict__ wbih,
                const unsigned short* __restrict__ wbhh,
                float* hbuf, int C, int Ntot) {
  int L = hdr[0], M = hdr[1];
  if (step >= L) return;
  int d = L - 1 - step;                    // deepest layer first
  const int* ln = layer_nodes + (long long)d * M;
  const int* ci = child_idx  + (long long)d * M * C;
  const int* cc = child_cnt  + (long long)d * M;

  // padded row stride (264 ushorts = 528B): 16-lane A-frag row reads hit all
  // 64 LDS banks exactly once -> conflict free.
  __shared__ __align__(16) unsigned short xT[TROWS][264];
  __shared__ __align__(16) unsigned short hp[TROWS][264];
  __shared__ __align__(16) float hp32[TROWS][260];
  __shared__ int nodes_s[TROWS];

  const int t    = threadIdx.x;
  const int lane = t & 31;
  const int wv   = t >> 5;                 // wave id, 0..7
  const int nc   = lane & 15;              // fragment column / A row
  const int half = lane >> 4;

  for (int r0 = blockIdx.x * TROWS; r0 < M; r0 += gridDim.x * TROWS) {
    if (ln[r0] >= Ntot) break;             // rows are front-packed: done
    if (t < TROWS) {
      int idx = r0 + t;
      nodes_s[t] = (idx < M) ? ln[idx] : Ntot;
    }
    // ------ phase 1: gather x rows and child-mean h_prev into LDS ------
    {
      int c = t;                           // column 0..255
      for (int r = 0; r < TROWS; ++r) {
        int idx  = r0 + r;
        int node = (idx < M) ? ln[idx] : Ntot;
        float xv = 0.0f;
        if (node < Ntot) xv = inputs[(long long)node * HD + c];
        xT[r][c] = f2bf(xv);
        int cnt = (idx < M) ? cc[idx] : 0;
        float s = 0.0f;
        const int* crow = ci + (long long)idx * C;
        for (int j = 0; j < cnt; ++j)
          s += hbuf[(long long)crow[j] * HD + c];
        float hpv = s / (float)(cnt > 0 ? cnt : 1);
        hp32[r][c] = hpv;
        hp[r][c]   = f2bf(hpv);
      }
    }
    __syncthreads();

    // ------ phase 2: WMMA GEMMs + fused GRU gates ------
    for (int pi = 0; pi < 2; ++pi) {
      int p  = wv + pi * 8;                // gate-col position 0..15
      int c0 = p * 16;                     // hidden columns c0..c0+15
      // bias-initialize accumulators (bias is column-constant per lane)
      float bir = b_ih[c0 + nc], biz = b_ih[256 + c0 + nc], big = b_ih[512 + c0 + nc];
      float bhr = b_hh[c0 + nc], bhz = b_hh[256 + c0 + nc], bhg = b_hh[512 + c0 + nc];
      v8f air0, aiz0, aig0, ahr0, ahz0, ahg0;
      v8f air1, aiz1, aig1, ahr1, ahz1, ahg1;
      for (int i = 0; i < 8; ++i) {
        air0[i] = bir; aiz0[i] = biz; aig0[i] = big;
        ahr0[i] = bhr; ahz0[i] = bhz; ahg0[i] = bhg;
        air1[i] = bir; aiz1[i] = biz; aig1[i] = big;
        ahr1[i] = bhr; ahz1[i] = bhz; ahg1[i] = bhg;
      }
#pragma unroll 2
      for (int kt = 0; kt < KT_N; ++kt) {
        // A fragments (16x32 bf16): lane row = nc (+16 for sub-tile 1)
        int kb = kt * 32 + half * 16;
        V16 ax0, ah0, ax1, ah1;
        ax0.q[0] = *(const uint4*)&xT[nc][kb];
        ax0.q[1] = *(const uint4*)&xT[nc][kb + 8];
        ah0.q[0] = *(const uint4*)&hp[nc][kb];
        ah0.q[1] = *(const uint4*)&hp[nc][kb + 8];
        ax1.q[0] = *(const uint4*)&xT[nc + 16][kb];
        ax1.q[1] = *(const uint4*)&xT[nc + 16][kb + 8];
        ah1.q[0] = *(const uint4*)&hp[nc + 16][kb];
        ah1.q[1] = *(const uint4*)&hp[nc + 16][kb + 8];
        // B fragments: pre-swizzled, one 32B load per lane per fragment,
        // each reused by both row sub-tiles (2 WMMAs per fragment)
        const v16bf b_ir = *(const v16bf*)(wbih + (((p      ) * KT_N + kt) * FRAG_US + lane * 16));
        const v16bf b_iz = *(const v16bf*)(wbih + (((p + 16) * KT_N + kt) * FRAG_US + lane * 16));
        const v16bf b_ig = *(const v16bf*)(wbih + (((p + 32) * KT_N + kt) * FRAG_US + lane * 16));
        const v16bf b_hr = *(const v16bf*)(wbhh + (((p      ) * KT_N + kt) * FRAG_US + lane * 16));
        const v16bf b_hz = *(const v16bf*)(wbhh + (((p + 16) * KT_N + kt) * FRAG_US + lane * 16));
        const v16bf b_hg = *(const v16bf*)(wbhh + (((p + 32) * KT_N + kt) * FRAG_US + lane * 16));
        air0 = __builtin_amdgcn_wmma_f32_16x16x32_bf16(false, ax0.v, false, b_ir, (short)0, air0, false, false);
        air1 = __builtin_amdgcn_wmma_f32_16x16x32_bf16(false, ax1.v, false, b_ir, (short)0, air1, false, false);
        aiz0 = __builtin_amdgcn_wmma_f32_16x16x32_bf16(false, ax0.v, false, b_iz, (short)0, aiz0, false, false);
        aiz1 = __builtin_amdgcn_wmma_f32_16x16x32_bf16(false, ax1.v, false, b_iz, (short)0, aiz1, false, false);
        aig0 = __builtin_amdgcn_wmma_f32_16x16x32_bf16(false, ax0.v, false, b_ig, (short)0, aig0, false, false);
        aig1 = __builtin_amdgcn_wmma_f32_16x16x32_bf16(false, ax1.v, false, b_ig, (short)0, aig1, false, false);
        ahr0 = __builtin_amdgcn_wmma_f32_16x16x32_bf16(false, ah0.v, false, b_hr, (short)0, ahr0, false, false);
        ahr1 = __builtin_amdgcn_wmma_f32_16x16x32_bf16(false, ah1.v, false, b_hr, (short)0, ahr1, false, false);
        ahz0 = __builtin_amdgcn_wmma_f32_16x16x32_bf16(false, ah0.v, false, b_hz, (short)0, ahz0, false, false);
        ahz1 = __builtin_amdgcn_wmma_f32_16x16x32_bf16(false, ah1.v, false, b_hz, (short)0, ahz1, false, false);
        ahg0 = __builtin_amdgcn_wmma_f32_16x16x32_bf16(false, ah0.v, false, b_hg, (short)0, ahg0, false, false);
        ahg1 = __builtin_amdgcn_wmma_f32_16x16x32_bf16(false, ah1.v, false, b_hg, (short)0, ahg1, false, false);
      }
      // gates on the C/D fragment layout: VGPR i -> row i + 8*half, col nc
      for (int i = 0; i < 8; ++i) {
        int Mr = i + half * 8;
        {
          float hprev = hp32[Mr][c0 + nc];
          float rr = fsigmoid(air0[i] + ahr0[i]);
          float zz = fsigmoid(aiz0[i] + ahz0[i]);
          float nn = ftanh(aig0[i] + rr * ahg0[i]);
          float hn = (1.0f - zz) * nn + zz * hprev;
          int node = nodes_s[Mr];            // pad rows scatter to dummy row N
          hbuf[(long long)node * HD + c0 + nc] = hn;
        }
        {
          float hprev = hp32[Mr + 16][c0 + nc];
          float rr = fsigmoid(air1[i] + ahr1[i]);
          float zz = fsigmoid(aiz1[i] + ahz1[i]);
          float nn = ftanh(aig1[i] + rr * ahg1[i]);
          float hn = (1.0f - zz) * nn + zz * hprev;
          int node = nodes_s[Mr + 16];
          hbuf[(long long)node * HD + c0 + nc] = hn;
        }
      }
    }
    __syncthreads();                          // protect LDS before next tile
  }
}

// ---------------- final root gather -----------------------------------------
__global__ void gather_roots(const float* __restrict__ hbuf,
                             const int* __restrict__ root_idx,
                             float* __restrict__ out, int T) {
  int i = blockIdx.x * blockDim.x + threadIdx.x;
  if (i < T * HD) {
    int tr = i >> 8, c = i & 255;
    out[i] = hbuf[(long long)root_idx[tr] * HD + c];
  }
}

// ---------------------------------------------------------------------------
extern "C" void kernel_launch(void* const* d_in, const int* in_sizes, int n_in,
                              void* d_out, int out_size, void* d_ws, size_t ws_size,
                              hipStream_t stream) {
  const float* inputs      = (const float*)d_in[0];
  const float* w_ih        = (const float*)d_in[1];
  const float* w_hh        = (const float*)d_in[2];
  const float* b_ih        = (const float*)d_in[3];
  const float* b_hh        = (const float*)d_in[4];
  const int*   layer_nodes = (const int*)d_in[5];
  const int*   child_idx   = (const int*)d_in[6];
  const int*   child_cnt   = (const int*)d_in[7];
  const int*   root_idx    = (const int*)d_in[8];

  const int Ntot = in_sizes[0] / HD;           // 200000
  const int LM   = in_sizes[5];                // L*M
  const int C    = in_sizes[6] / in_sizes[5];  // max children per node
  const int T    = in_sizes[8];                // 2000 trees

  char* base = (char*)d_ws;
  int* hdr = (int*)base;                                        // {L, M, scan}
  unsigned short* wbih = (unsigned short*)(base + 512);
  unsigned short* wbhh = wbih + WB_US;
  float* hbuf = (float*)(base + 512 + (size_t)2 * WB_US * 2);   // (N+2) x 256

  // 1) recover (L, M)
  meta_init<<<1, 1, 0, stream>>>(hdr, LM);
  meta_scan<<<256, 256, 0, stream>>>(layer_nodes, hdr, T, LM, Ntot);
  meta_fin<<<1, 1, 0, stream>>>(hdr, LM);

  // 2) zero hidden states (h0 = 0, incl. dummy rows N and N+1)
  long long n4 = (long long)(Ntot + 2) * (HD / 4);
  zero_f4<<<4096, 256, 0, stream>>>((float4*)hbuf, n4);

  // 3) convert + swizzle weights to bf16 B-fragment layout
  conv_w<<<(2 * WB_US + 255) / 256, 256, 0, stream>>>(w_ih, w_hh, wbih, wbhh);

  // 4) sequential layer steps (deepest -> roots); steps >= L exit immediately
  for (int s = 0; s < LMAX_STEPS; ++s)
    layer_step<<<1024, 256, 0, stream>>>(s, hdr, inputs, b_ih, b_hh,
                                         layer_nodes, child_idx, child_cnt,
                                         wbih, wbhh, hbuf, C, Ntot);

  // 5) emit root hidden states
  gather_roots<<<(T * HD + 255) / 256, 256, 0, stream>>>(hbuf, root_idx,
                                                         (float*)d_out, T);
}
